// NodeGraphContrastiveLoss_9955734192142
// MI455X (gfx1250) — matrix operand order repaired and checked by reading, not compiled, not run on hardware
//
#include <hip/hip_runtime.h>
#include <hip/hip_bf16.h>

typedef __attribute__((ext_vector_type(16))) _Float16 v16h;
typedef __attribute__((ext_vector_type(8)))  _Float16 v8h;
typedef __attribute__((ext_vector_type(8)))  float    v8f;

#define N_ROWS   131072          // 1024 * 128
#define C_DIM    256
#define K_GRAPH  1024
#define INV_T    5.0f            // 1 / 0.2

#if defined(__HIP_DEVICE_COMPILE__) && __has_builtin(__builtin_amdgcn_global_load_async_to_lds_b128)
#define USE_ASYNC_LDS 1
typedef int v4i_vs __attribute__((vector_size(16)));
typedef __attribute__((address_space(1))) v4i_vs as1_v4i;   // global (prints as __device__)
typedef __attribute__((address_space(3))) v4i_vs as3_v4i;   // LDS
#else
#define USE_ASYNC_LDS 0
#endif

__device__ __forceinline__ void wait_async_lds() {
#if USE_ASYNC_LDS
#if __has_builtin(__builtin_amdgcn_s_wait_asynccnt)
    __builtin_amdgcn_s_wait_asynccnt(0);
#else
    asm volatile("s_wait_asynccnt 0x0" ::: "memory");
#endif
#endif
}

// Stage one 128-row G chunk (128*256 f16 = 64 KB) into an LDS buffer.
// Async path: global_load_async_to_lds_b128 (ASYNCcnt), no VGPR round-trip.
// Fallback: plain b128 load + ds store (consumed only after next barrier).
__device__ __forceinline__ void stage_g_chunk(const _Float16* __restrict__ gsrc,
                                              _Float16* __restrict__ dst, int tid) {
#if USE_ASYNC_LDS
#pragma unroll
    for (int j = 0; j < 16; ++j) {               // 4096 x 16B transfers / 256 threads
        int i = tid + j * 256;
        __builtin_amdgcn_global_load_async_to_lds_b128(
            (as1_v4i*)(gsrc + (size_t)i * 8),
            (as3_v4i*)(dst + (size_t)i * 8), 0, 0);
    }
#else
    const float4* s = (const float4*)gsrc;
    float4* d = (float4*)dst;
    for (int i = tid; i < 4096; i += 256) d[i] = s[i];
#endif
}

// ---------------- Kernel 1: inverse L2 norm of each l_enc row (1 wave / row) ----
__global__ __launch_bounds__(256) void lnorm_kernel(const float* __restrict__ L,
                                                    float* __restrict__ xinv) {
    int gwave = (blockIdx.x * 256 + threadIdx.x) >> 5;   // 0 .. N_ROWS-1
    int lane  = threadIdx.x & 31;
    const float4* r4 = (const float4*)(L + (size_t)gwave * C_DIM);
    float4 a = r4[lane];
    float4 b = r4[lane + 32];
    float s = a.x*a.x + a.y*a.y + a.z*a.z + a.w*a.w
            + b.x*b.x + b.y*b.y + b.z*b.z + b.w*b.w;
#pragma unroll
    for (int off = 1; off < 32; off <<= 1) s += __shfl_xor(s, off, 32);
    if (lane == 0) xinv[gwave] = rsqrtf(s);
}

// ---------------- Kernel 2: normalize g_enc rows, convert to f16 ----------------
__global__ __launch_bounds__(256) void gprep_kernel(const float* __restrict__ G,
                                                    _Float16* __restrict__ Gh) {
    int gwave = (blockIdx.x * 256 + threadIdx.x) >> 5;   // 0 .. K_GRAPH-1
    int lane  = threadIdx.x & 31;
    const float4* r4 = (const float4*)(G + (size_t)gwave * C_DIM);
    float4 a = r4[lane];
    float4 b = r4[lane + 32];
    float s = a.x*a.x + a.y*a.y + a.z*a.z + a.w*a.w
            + b.x*b.x + b.y*b.y + b.z*b.z + b.w*b.w;
#pragma unroll
    for (int off = 1; off < 32; off <<= 1) s += __shfl_xor(s, off, 32);
    float inv = rsqrtf(s);
    _Float16* dst = Gh + (size_t)gwave * C_DIM;
    int p0 = 4 * lane;
    dst[p0+0]       = (_Float16)(a.x * inv);
    dst[p0+1]       = (_Float16)(a.y * inv);
    dst[p0+2]       = (_Float16)(a.z * inv);
    dst[p0+3]       = (_Float16)(a.w * inv);
    dst[p0+128+0]   = (_Float16)(b.x * inv);
    dst[p0+128+1]   = (_Float16)(b.y * inv);
    dst[p0+128+2]   = (_Float16)(b.z * inv);
    dst[p0+128+3]   = (_Float16)(b.w * inv);
}

// ---------------- Kernel 3: zero scalar output ----------------------------------
__global__ void zero_kernel(float* out) {
    if (threadIdx.x == 0) out[0] = 0.0f;
}

// ---------------- Kernel 4: WMMA similarity + contrastive reduction -------------
// grid = 1024 blocks (one per batch element b), 256 threads = 8 waves.
// LDS: 64 KB L strip + 2 x 64 KB double-buffered G chunks = 192 KB.
__global__ __launch_bounds__(256) void main_kernel(const float* __restrict__ L,
                                                   const float* __restrict__ xinv,
                                                   const _Float16* __restrict__ Gh,
                                                   float* __restrict__ out) {
    extern __shared__ char smem_raw[];
    _Float16* sL  = (_Float16*)smem_raw;          // 128 x 256 f16 = 64 KB
    _Float16* sG0 = sL + 128 * C_DIM;             // 64 KB
    _Float16* sG1 = sG0 + 128 * C_DIM;            // 64 KB

    const int b    = blockIdx.x;
    const int tid  = threadIdx.x;
    const int wave = tid >> 5;
    const int lane = tid & 31;
    const int lrow = lane & 15;
    const int hi   = lane >> 4;                   // 0: lanes 0-15, 1: lanes 16-31
    const int m0   = wave * 16;                   // 16-row strip per wave

    // ---- Prefetch G chunk 0 (async when available) ----
    stage_g_chunk(Gh, sG0, tid);

    // ---- Phase A: load L strip, normalize, convert to f16 into LDS ----
    {
        const float4* src = (const float4*)(L + (size_t)b * 128 * C_DIM);
        for (int i = tid; i < 128 * 64; i += 256) {      // 8192 float4's
            int row = i >> 6;
            float s = xinv[b * 128 + row];
            float4 v = src[i];
            _Float16* d = &sL[i * 4];
            d[0] = (_Float16)(v.x * s);
            d[1] = (_Float16)(v.y * s);
            d[2] = (_Float16)(v.z * s);
            d[3] = (_Float16)(v.w * s);
        }
    }

    // per-lane accumulators: rows (m0 + i + 8*hi), 8 rows per lane
    float racc[8] = {0.f, 0.f, 0.f, 0.f, 0.f, 0.f, 0.f, 0.f};
    float posc[8] = {0.f, 0.f, 0.f, 0.f, 0.f, 0.f, 0.f, 0.f};
    const int nt_pos = b >> 4;      // n-tile containing positive column k=b
    const int c_pos  = b & 15;      // column within tile

    const int kloA = hi * 16;       // A-operand per-lane K base (contiguous packing)
    const int kloB = hi * 16;       // B-operand per-lane K base (contiguous packing)
    const _Float16* arow = &sL[(m0 + lrow) * C_DIM];

    // ---- Phase B: 8 G-chunks of 128 rows, double-buffered, 8 n-tiles each ----
    for (int ch = 0; ch < 8; ++ch) {
        _Float16* cur = (ch & 1) ? sG1 : sG0;
        _Float16* nxt = (ch & 1) ? sG0 : sG1;

        wait_async_lds();           // this wave's async fills of `cur` complete
        __syncthreads();            // all waves done: chunk ch visible; `nxt` readers done

        if (ch < 7) stage_g_chunk(Gh + (size_t)(ch + 1) * 128 * C_DIM, nxt, tid);

        for (int t = 0; t < 8; ++t) {
            const int nt = ch * 8 + t;
            const _Float16* brow = &cur[(t * 16 + lrow) * C_DIM];
            v8f acc = {};
#pragma unroll
            for (int kc = 0; kc < 8; ++kc) {
                const int ka = kc * 32 + kloA;
                const int kb = kc * 32 + kloB;
                v8h alo = *(const v8h*)(arow + ka);
                v8h ahi = *(const v8h*)(arow + ka + 8);
                v8h blo = *(const v8h*)(brow + kb);
                v8h bhi = *(const v8h*)(brow + kb + 8);
                v16h av = __builtin_shufflevector(alo, ahi, 0,1,2,3,4,5,6,7,8,9,10,11,12,13,14,15);
                v16h bv = __builtin_shufflevector(blo, bhi, 0,1,2,3,4,5,6,7,8,9,10,11,12,13,14,15);
                acc = __builtin_amdgcn_wmma_f32_16x16x32_f16(
                          false, av, false, bv, (short)0, acc, false, false);
            }
            const bool is_pos = (nt == nt_pos) && (lrow == c_pos);
#pragma unroll
            for (int i = 0; i < 8; ++i) {
                float c = acc[i];                 // cos sim, row m0+i+8*hi, col nt*16+lrow
                racc[i] += __expf(c * INV_T);
                if (is_pos) posc[i] = c;
            }
        }
    }

    // ---- Phase C: reduce row sums across the 16 lanes of each half-wave ----
#pragma unroll
    for (int off = 1; off < 16; off <<= 1) {
#pragma unroll
        for (int i = 0; i < 8; ++i) racc[i] += __shfl_xor(racc[i], off, 32);
    }

    // lanes (c_pos) and (c_pos+16) hold the positive cosines for their 8 rows
    if (lrow == c_pos) {
        float local = 0.f;
#pragma unroll
        for (int i = 0; i < 8; ++i) {
            float e = __expf(posc[i] * INV_T);
            // -log(l_pos / (sum - l_pos)) = log(sum - l_pos) - cos/T
            local += __logf(racc[i] - e) - posc[i] * INV_T;
        }
        atomicAdd(out, local * (1.0f / (float)N_ROWS));
    }
}

extern "C" void kernel_launch(void* const* d_in, const int* in_sizes, int n_in,
                              void* d_out, int out_size, void* d_ws, size_t ws_size,
                              hipStream_t stream) {
    const float* L = (const float*)d_in[0];     // l_enc [1024,128,256] f32
    const float* G = (const float*)d_in[1];     // g_enc [1024,256] f32
    float* out = (float*)d_out;

    float*    xinv = (float*)d_ws;                                     // 512 KB
    _Float16* Gh   = (_Float16*)((char*)d_ws + (size_t)N_ROWS * 4);    // 512 KB

    lnorm_kernel<<<N_ROWS / 8, 256, 0, stream>>>(L, xinv);             // 8 waves/block
    gprep_kernel<<<K_GRAPH / 8, 256, 0, stream>>>(G, Gh);
    zero_kernel<<<1, 32, 0, stream>>>(out);
    main_kernel<<<1024, 256, 192 * 1024, stream>>>(L, xinv, Gh, out);  // 192 KB LDS
}
